// GATAttention_1597727834390
// MI455X (gfx1250) — matrix operand order repaired and checked by reading, not compiled, not run on hardware
//
#include <hip/hip_runtime.h>
#include <hip/hip_bf16.h>
#include <math.h>

typedef __attribute__((ext_vector_type(16))) __bf16 v16bf;
typedef __attribute__((ext_vector_type(8)))  __bf16 v8bf;
typedef __attribute__((ext_vector_type(8)))  float  v8f;
typedef __attribute__((ext_vector_type(4)))  unsigned int u32x4;
typedef __attribute__((ext_vector_type(8)))  int i32x8;
typedef __attribute__((ext_vector_type(4)))  int i32x4;

#define N_NODES 4096
#define IN_DIM  256
#define HID     64
#define HEADS   8
#define DMODEL  (HID * HEADS)   // 512

#if defined(__has_builtin)
#if __has_builtin(__builtin_amdgcn_tensor_load_to_lds) && __has_builtin(__builtin_amdgcn_s_wait_tensorcnt)
#define HAS_TDM 1
#endif
#endif
#ifndef HAS_TDM
#define HAS_TDM 0
#endif

#if HAS_TDM
#if __clang_major__ >= 23
#warning "CDNA5 probe: TDM path enabled (6-arg tensor_load_to_lds, clang>=23)"
#else
#warning "CDNA5 probe: TDM path enabled (5-arg tensor_load_to_lds, clang<=22)"
#endif
#else
#warning "CDNA5 probe: tensor_load_to_lds builtin absent; cooperative LDS copy fallback"
#endif

// ---------------- WMMA fragment helpers (contiguous k order: element j -> k = 16*half + j) ------

__device__ __forceinline__ v8f wmma_bf16(v16bf a, v16bf b, v8f c) {
  return __builtin_amdgcn_wmma_f32_16x16x32_bf16(false, a, false, b,
                                                 (short)0, c, false, false);
}

// A-matrix 16x32 bf16: lane holds row (lane&15); element j -> K offset
__device__ __forceinline__ int a_koff(int j, int half) {
  return half * 16 + j;
}

__device__ __forceinline__ v16bf a_frag_from_f32(const float* row, int k0, int half) {
  v16bf a;
#pragma unroll
  for (int j = 0; j < 16; ++j) a[j] = (__bf16)row[k0 + a_koff(j, half)];
  return a;
}

// vectorized: two contiguous 16B halves (16B aligned at all call sites)
__device__ __forceinline__ v16bf a_frag_from_bf16(const __bf16* row, int k0, int half) {
  v8bf lo = *(const v8bf*)(row + k0 + half * 16);
  v8bf hi = *(const v8bf*)(row + k0 + half * 16 + 8);
  v16bf a;
#pragma unroll
  for (int j = 0; j < 8; ++j) { a[j] = lo[j]; a[j + 8] = hi[j]; }
  return a;
}

// B-matrix 32x16: lane holds column (lane&15); element j -> K = half*16 + j
__device__ __forceinline__ v16bf b_frag_from_f32(const float* base, int ld,
                                                 int k0, int col, int half) {
  v16bf b;
#pragma unroll
  for (int j = 0; j < 16; ++j)
    b[j] = (__bf16)base[(size_t)(k0 + half * 16 + j) * ld + col];
  return b;
}

// B = X^T with X row-major: contiguous along k -> single 32B vector load
// (32B aligned at all call sites: bases are multiples of 128B, k0*2 in {0,64}B,
//  half*16*2 = 32B)
__device__ __forceinline__ v16bf b_frag_transposed_bf16(const __bf16* xrow,
                                                        int k0, int half) {
  return *(const v16bf*)(xrow + k0 + half * 16);
}

__device__ __forceinline__ float gelu_exact(float x) {
  return 0.5f * x * (1.0f + erff(x * 0.70710678118654752f));
}

__device__ __forceinline__ unsigned int lds_addr_of(const void* p) {
  // generic ptr to a __shared__ object: low 32 bits are the LDS byte address
  return (unsigned int)(unsigned long long)(uintptr_t)p;
}

// ---- TDM: async DMA of a 32-row x 64-col bf16 tile (row stride DMODEL) to LDS
__device__ __forceinline__ void tdm_load_tile(const __bf16* gsrc, unsigned int lds_addr) {
#if HAS_TDM
  unsigned long long ga = (unsigned long long)(uintptr_t)gsrc;
  u32x4 g0 = {};
  g0[0] = 1u;                                   // count=1, user mode, no gather
  g0[1] = lds_addr;                             // [63:32]  lds_addr
  g0[2] = (unsigned int)ga;                     // [95:64]  global_addr[31:0]
  g0[3] = (unsigned int)((ga >> 32) & 0x01FFFFFFu) | (2u << 30); // addr[56:32], type=2
  i32x8 g1 = {};
  g1[0] = (int)(1u << 16);                      // wg_mask=0, data_size=1 (2 bytes)
  g1[1] = (int)((DMODEL & 0xFFFFu) << 16);      // tensor_dim0[15:0] at bits 63:48
  g1[2] = (int)(((unsigned)DMODEL >> 16) | ((N_NODES & 0xFFFFu) << 16)); // dim0 hi, dim1 lo
  g1[3] = (int)(((unsigned)N_NODES >> 16) | ((unsigned)HID << 16));      // dim1 hi, tile_dim0=64
  g1[4] = (int)32;                              // tile_dim1=32, tile_dim2=0
  g1[5] = (int)DMODEL;                          // tensor_dim0_stride = 512
  g1[6] = 0; g1[7] = 0;                         // dim1_stride unused (2-D)
  i32x4 z4 = {};
#if __clang_major__ >= 23
  i32x8 z8 = {};
  __builtin_amdgcn_tensor_load_to_lds(g0, g1, z4, z4, z8, 0);
#else
  __builtin_amdgcn_tensor_load_to_lds(g0, g1, z4, z4, 0);
#endif
#else
  (void)gsrc; (void)lds_addr;
#endif
}

// fallback cooperative copy: 32 rows x 64 bf16 (16B chunks)
__device__ __forceinline__ void copy_tile_fallback(const __bf16* gsrc, __bf16* dst, int lane) {
#pragma unroll
  for (int c = lane; c < 256; c += 32) {
    int r = c >> 3, o = (c & 7) * 8;
    *(uint4*)(dst + r * HID + o) = *(const uint4*)(gsrc + (size_t)r * DMODEL + o);
  }
}

// ---------------- 1) proj = features @ W_in   (fp32 in, bf16 out) ----------
// block = one 16-col tile of W_in; the 256x16 panel is staged once in LDS as
// bf16 (contiguous 64B row reads), 4 waves compute 4 row tiles against it.

__global__ void __launch_bounds__(128)
k_proj(const float* __restrict__ feats, const float* __restrict__ Win,
       __bf16* __restrict__ proj) {
  __shared__ __bf16 Wl[IN_DIM * 16];   // 8KB W_in panel (K=256, 16 cols)

  int tid  = threadIdx.x;
  int wave = tid >> 5, lane = tid & 31;
  int half = lane >> 4, nn = lane & 15;
  int ct = blockIdx.x & 31;            // 32 col tiles
  int rg = blockIdx.x >> 5;            // 64 groups of 4 row tiles
  int col0 = ct * 16;

  // stage W_in[:, col0:col0+16] -> LDS bf16 (vectorized 32B row stores)
  for (int r = tid; r < IN_DIM; r += 128) {
    const float* src = Win + (size_t)r * DMODEL + col0;
    v16bf tmp;
#pragma unroll
    for (int c = 0; c < 16; ++c) tmp[c] = (__bf16)src[c];
    *(v16bf*)(Wl + r * 16) = tmp;
  }
  __syncthreads();

  int row0 = (rg * 4 + wave) * 16;
  const float* arow = feats + (size_t)(row0 + nn) * IN_DIM;
  v8f acc = {};
#pragma unroll
  for (int k0 = 0; k0 < IN_DIM; k0 += 32) {
    v16bf a = a_frag_from_f32(arow, k0, half);
    v16bf b;
#pragma unroll
    for (int j = 0; j < 16; ++j) b[j] = Wl[(k0 + half * 16 + j) * 16 + nn];
    acc = wmma_bf16(a, b, acc);
  }
#pragma unroll
  for (int r = 0; r < 8; ++r) {
    int row = row0 + r + 8 * half;
    proj[(size_t)row * DMODEL + col0 + nn] = (__bf16)acc[r];
  }
}

// ---------------- 2) flash-style masked attention per head ------------------
// one wave per (16 query rows, head); 32-key chunks; K/V tile double-buffered
// in LDS and fed by the Tensor Data Mover (async, s_wait_tensorcnt).

__global__ void __launch_bounds__(32)
k_attn(const __bf16* __restrict__ proj, const unsigned char* __restrict__ adj,
       __bf16* __restrict__ attended) {
  __shared__ __bf16 Vl[2][32 * HID];   // 2 x 4KB K/V tile (32 keys x 64 dims)
  __shared__ __bf16 Pl[16 * 32];       // P relayout buffer

  int wid  = blockIdx.x;       // 256 row tiles * 8 heads
  int h    = wid & 7;
  int rt   = wid >> 3;
  int lane = threadIdx.x & 31;
  int half = lane >> 4, nn = lane & 15;
  int row0 = rt * 16;
  int hoff = h * HID;

  const __bf16* qrow = proj + (size_t)(row0 + nn) * DMODEL + hoff;
  v16bf qa0 = a_frag_from_bf16(qrow, 0, half);
  v16bf qa1 = a_frag_from_bf16(qrow, 32, half);

  float m[8], l[8];
  v8f zero = {};
  v8f o[4] = {zero, zero, zero, zero};
#pragma unroll
  for (int r = 0; r < 8; ++r) { m[r] = -__builtin_inff(); l[r] = 0.0f; }

  // prime the pipeline: DMA first K/V tile into buffer 0
#if HAS_TDM
  tdm_load_tile(proj + hoff, lds_addr_of(&Vl[0][0]));
#else
  copy_tile_fallback(proj + hoff, &Vl[0][0], lane);
  __syncthreads();
#endif

  for (int kb = 0; kb < N_NODES; kb += 32) {
    int cur = (kb >> 5) & 1;

    // overlap: DMA the next tile into the other buffer while computing this one
    if (kb + 32 < N_NODES) {
#if HAS_TDM
      tdm_load_tile(proj + (size_t)(kb + 32) * DMODEL + hoff,
                    lds_addr_of(&Vl[cur ^ 1][0]));
      __builtin_amdgcn_s_wait_tensorcnt(1);   // current tile landed (in-order)
#else
      copy_tile_fallback(proj + (size_t)(kb + 32) * DMODEL + hoff,
                         &Vl[cur ^ 1][0], lane);
      __syncthreads();
#endif
    } else {
#if HAS_TDM
      __builtin_amdgcn_s_wait_tensorcnt(0);
#endif
    }

    // warm next adjacency cachelines through L2 (global_prefetch_b8)
    __builtin_prefetch(adj + (size_t)(row0 + nn) * N_NODES + kb + 32, 0, 0);

    const __bf16* Vc = &Vl[cur][0];

    // S = Q K^T  (two 16x16 tiles, K=64) — K rows read from LDS tile (b128)
    v8f s[2];
#pragma unroll
    for (int t = 0; t < 2; ++t) {
      const __bf16* krow = Vc + (t * 16 + nn) * HID;
      v16bf b0 = b_frag_transposed_bf16(krow, 0, half);
      v8f st = wmma_bf16(qa0, b0, zero);
      v16bf b1 = b_frag_transposed_bf16(krow, 32, half);
      st = wmma_bf16(qa1, b1, st);
      s[t] = st;
    }

    // scale + adjacency mask + online softmax (row = r + 8*half, col = lane&15)
    float mnew[8];
#pragma unroll
    for (int r = 0; r < 8; ++r) {
      int row = row0 + r + 8 * half;
#pragma unroll
      for (int t = 0; t < 2; ++t) {
        int col = kb + t * 16 + nn;
        float v = s[t][r] * 0.125f;   // 1/sqrt(64)
        s[t][r] = adj[(size_t)row * N_NODES + col] ? v : -__builtin_inff();
      }
      float rv = fmaxf(s[0][r], s[1][r]);
#pragma unroll
      for (int off = 1; off < 16; off <<= 1)
        rv = fmaxf(rv, __shfl_xor(rv, off, 32));
      mnew[r] = fmaxf(m[r], rv);
    }

#pragma unroll
    for (int r = 0; r < 8; ++r) {
      bool dead = (mnew[r] == -__builtin_inff());
      float mref = dead ? 0.0f : mnew[r];
      float p0 = __expf(s[0][r] - mref);
      float p1 = __expf(s[1][r] - mref);
      s[0][r] = p0; s[1][r] = p1;
      float rs = p0 + p1;
#pragma unroll
      for (int off = 1; off < 16; off <<= 1)
        rs += __shfl_xor(rs, off, 32);
      float f = dead ? 1.0f : __expf(m[r] - mnew[r]);
      l[r] = l[r] * f + rs;
      m[r] = mnew[r];
#pragma unroll
      for (int t = 0; t < 4; ++t) o[t][r] *= f;
    }

    // relayout P (C layout -> A layout) through LDS; 1-wave WG: barrier == S_NOP
    __syncthreads();
#pragma unroll
    for (int r = 0; r < 8; ++r) {
      int prow = r + 8 * half;
      Pl[prow * 32 + nn]      = (__bf16)s[0][r];
      Pl[prow * 32 + 16 + nn] = (__bf16)s[1][r];
    }
    __syncthreads();
    v16bf pa = a_frag_from_bf16(Pl + nn * 32, 0, half);

    // O += P @ V   (V rows from the LDS tile, 4 column tiles of 16)
#pragma unroll
    for (int t = 0; t < 4; ++t) {
      v16bf vb;
#pragma unroll
      for (int j = 0; j < 16; ++j)
        vb[j] = Vc[(half * 16 + j) * HID + t * 16 + nn];
      o[t] = wmma_bf16(pa, vb, o[t]);
    }
  }

#pragma unroll
  for (int r = 0; r < 8; ++r) {
    float inv = 1.0f / fmaxf(l[r], 1e-20f);
    int row = row0 + r + 8 * half;
#pragma unroll
    for (int t = 0; t < 4; ++t)
      attended[(size_t)row * DMODEL + hoff + t * 16 + nn] = (__bf16)(o[t][r] * inv);
  }
}

// ---------------- 3) node_states = attended @ W_out + b_out ----------------

__global__ void __launch_bounds__(128)
k_outproj(const __bf16* __restrict__ attended, const float* __restrict__ Wout,
          const float* __restrict__ bout, float* __restrict__ ns) {
  int wid  = (blockIdx.x * blockDim.x + threadIdx.x) >> 5;
  int lane = threadIdx.x & 31;
  int half = lane >> 4, nn = lane & 15;
  int rt = wid >> 2, ct = wid & 3;     // 256 x 4 tiles
  int row0 = rt * 16, col0 = ct * 16;
  const __bf16* arow = attended + (size_t)(row0 + nn) * DMODEL;
  v8f acc = {};
#pragma unroll
  for (int k0 = 0; k0 < DMODEL; k0 += 32) {
    v16bf a = a_frag_from_bf16(arow, k0, half);
    v16bf b = b_frag_from_f32(Wout, HID, k0, col0 + nn, half);
    acc = wmma_bf16(a, b, acc);
  }
  float bias = bout[col0 + nn];
#pragma unroll
  for (int r = 0; r < 8; ++r)
    ns[(size_t)(row0 + r + 8 * half) * HID + col0 + nn] = acc[r] + bias;
}

// ---- 4) h = gelu(node_states) @ W_c1[:64] + b_c1  (zeros half contributes 0)

__global__ void __launch_bounds__(128)
k_h(const float* __restrict__ ns, const float* __restrict__ Wc1,
    const float* __restrict__ bc1, float* __restrict__ hbuf) {
  int wid  = (blockIdx.x * blockDim.x + threadIdx.x) >> 5;
  int lane = threadIdx.x & 31;
  int half = lane >> 4, nn = lane & 15;
  int rt = wid >> 2, ct = wid & 3;
  int row0 = rt * 16, col0 = ct * 16;
  const float* arow = ns + (size_t)(row0 + nn) * HID;
  v8f acc = {};
#pragma unroll
  for (int k0 = 0; k0 < HID; k0 += 32) {
    v16bf a;
#pragma unroll
    for (int j = 0; j < 16; ++j)
      a[j] = (__bf16)gelu_exact(arow[k0 + a_koff(j, half)]);
    v16bf b = b_frag_from_f32(Wc1, HID, k0, col0 + nn, half);
    acc = wmma_bf16(a, b, acc);
  }
  float bias = bc1[col0 + nn];
#pragma unroll
  for (int r = 0; r < 8; ++r)
    hbuf[(size_t)(row0 + r + 8 * half) * HID + col0 + nn] = acc[r] + bias;
}

// ---------------- 5) scores = gelu(h) @ W_c2 + b_c2 ------------------------

__global__ void k_scores(const float* __restrict__ hbuf, const float* __restrict__ Wc2,
                         const float* __restrict__ bc2, float* __restrict__ scores) {
  int n = blockIdx.x * blockDim.x + threadIdx.x;
  if (n >= N_NODES) return;
  float s = bc2[0];
#pragma unroll 8
  for (int d = 0; d < HID; ++d)
    s += gelu_exact(hbuf[(size_t)n * HID + d]) * Wc2[d];
  scores[n] = s;
}

// ---------------- 6) xn = node_states / max(||ns||, eps)  (bf16) -----------

__global__ void k_xn(const float* __restrict__ ns, __bf16* __restrict__ xn) {
  int n = blockIdx.x * blockDim.x + threadIdx.x;
  if (n >= N_NODES) return;
  float ss = 0.0f;
#pragma unroll 8
  for (int d = 0; d < HID; ++d) { float v = ns[(size_t)n * HID + d]; ss += v * v; }
  float inv = 1.0f / fmaxf(sqrtf(ss), 1e-8f);
#pragma unroll 8
  for (int d = 0; d < HID; ++d)
    xn[(size_t)n * HID + d] = (__bf16)(ns[(size_t)n * HID + d] * inv);
}

__global__ void k_zero(float* __restrict__ p) {
  int i = blockIdx.x * blockDim.x + threadIdx.x;
  if (i < N_NODES) p[i] = 0.0f;
}

// ---------------- 7) sim = mask(xn @ xn^T), row-sums via atomics ------------

__global__ void __launch_bounds__(128)
k_sim(const __bf16* __restrict__ xn, const unsigned char* __restrict__ adj,
      float* __restrict__ heat, float* __restrict__ rowsum) {
  int wid  = (blockIdx.x * blockDim.x + threadIdx.x) >> 5;
  int lane = threadIdx.x & 31;
  int half = lane >> 4, nn = lane & 15;
  int rt = wid >> 8, ct = wid & 255;
  int row0 = rt * 16, col0 = ct * 16;
  const __bf16* arow = xn + (size_t)(row0 + nn) * HID;
  v16bf a0 = a_frag_from_bf16(arow, 0, half);
  v16bf a1 = a_frag_from_bf16(arow, 32, half);
  const __bf16* brow = xn + (size_t)(col0 + nn) * HID;
  v16bf b0 = b_frag_transposed_bf16(brow, 0, half);
  v16bf b1 = b_frag_transposed_bf16(brow, 32, half);
  v8f zero = {};
  v8f acc = wmma_bf16(a0, b0, zero);
  acc = wmma_bf16(a1, b1, acc);
#pragma unroll
  for (int r = 0; r < 8; ++r) {
    int row = row0 + r + 8 * half, col = col0 + nn;
    float v = adj[(size_t)row * N_NODES + col] ? acc[r] : 0.0f;
    heat[(size_t)row * N_NODES + col] = v;
    float rs = v;
#pragma unroll
    for (int off = 1; off < 16; off <<= 1) rs += __shfl_xor(rs, off, 32);
    if (nn == 0) atomicAdd(&rowsum[row], rs);
  }
}

// ---------------- 8) heatmap = sim / (rowsum + eps) -------------------------

__global__ void k_scale(float* __restrict__ heat, const float* __restrict__ rowsum) {
  size_t i = (size_t)blockIdx.x * blockDim.x + threadIdx.x;
  if (i >= (size_t)N_NODES * N_NODES) return;
  int row = (int)(i >> 12);   // / 4096
  heat[i] = heat[i] / (rowsum[row] + 1e-8f);
}

// ---------------- launcher --------------------------------------------------

extern "C" void kernel_launch(void* const* d_in, const int* in_sizes, int n_in,
                              void* d_out, int out_size, void* d_ws, size_t ws_size,
                              hipStream_t stream) {
  const float*         feats = (const float*)d_in[0];
  const unsigned char* adj   = (const unsigned char*)d_in[1];
  const float*         Win   = (const float*)d_in[2];
  const float*         Wout  = (const float*)d_in[3];
  const float*         bout  = (const float*)d_in[4];
  const float*         Wc1   = (const float*)d_in[5];
  const float*         bc1   = (const float*)d_in[6];
  const float*         Wc2   = (const float*)d_in[7];
  const float*         bc2   = (const float*)d_in[8];

  float* scores = (float*)d_out;
  float* heat   = scores + N_NODES;

  char* ws = (char*)d_ws;
  __bf16* proj     = (__bf16*)ws; ws += (size_t)N_NODES * DMODEL * sizeof(__bf16);
  __bf16* attended = (__bf16*)ws; ws += (size_t)N_NODES * DMODEL * sizeof(__bf16);
  float*  ns       = (float*) ws; ws += (size_t)N_NODES * HID * sizeof(float);
  float*  hbuf     = (float*) ws; ws += (size_t)N_NODES * HID * sizeof(float);
  __bf16* xn       = (__bf16*)ws; ws += (size_t)N_NODES * HID * sizeof(__bf16);
  float*  rowsum   = (float*) ws; ws += (size_t)N_NODES * sizeof(float);

  k_proj   <<<2048, 128, 0, stream>>>(feats, Win, proj);    // 32 ct x 64 row groups
  k_attn   <<<2048,  32, 0, stream>>>(proj, adj, attended); // 256 rt * 8 heads
  k_outproj<<< 256, 128, 0, stream>>>(attended, Wout, bout, ns);
  k_h      <<< 256, 128, 0, stream>>>(ns, Wc1, bc1, hbuf);
  k_scores <<<  16, 256, 0, stream>>>(hbuf, Wc2, bc2, scores);
  k_xn     <<<  16, 256, 0, stream>>>(ns, xn);
  k_zero   <<<  16, 256, 0, stream>>>(rowsum);
  k_sim    <<<16384,128, 0, stream>>>(xn, adj, heat, rowsum);
  k_scale  <<<65536,256, 0, stream>>>(heat, rowsum);
}
